// GraphLinkPredictor_54443005444448
// MI455X (gfx1250) — matrix operand order repaired and checked by reading, not compiled, not run on hardware
//
#include <hip/hip_runtime.h>
#include <hip/hip_bf16.h>
#include <math.h>

#define N_NODES  50000
#define N_EDGES  1600000
#define N_PAIRS  2000000
#define N_GRAPHS 50
#define HEADS    4
#define HID      32
#define F        128           // HEADS*HID
#define F2       256           // fused xl|xr width
#define K0       384           // layer-0 input width
#define ETOT     (N_EDGES + N_NODES)
#define NEG_SLOPE 0.2f
#define LN_EPS    1e-5f

typedef __attribute__((ext_vector_type(16))) _Float16 v16h;
typedef __attribute__((ext_vector_type(8)))  _Float16 v8h;
typedef __attribute__((ext_vector_type(8)))  float    v8f;

__device__ __forceinline__ float lrelu(float x) { return x > 0.f ? x : NEG_SLOPE * x; }

// monotonic float atomic-max via int/uint ordering trick
__device__ __forceinline__ void atomicMaxF(float* addr, float v) {
  if (v >= 0.f) atomicMax((int*)addr, __float_as_int(v));
  else          atomicMin((unsigned int*)addr, __float_as_uint(v));
}

// virtual self-loop append: edges [0,E) are real, [E, E+N) are loops
__device__ __forceinline__ void edge_sd(const int* __restrict__ ei, int e, int& s, int& d) {
  if (e < N_EDGES) { s = ei[e]; d = ei[N_EDGES + e]; }
  else             { s = e - N_EDGES; d = s; }
}

// ---------------- time embedding: h16[n, 0:128]=x, h16[n,128:384]=swish(feats@tW+tb)
__global__ void time_embed_kernel(const float* __restrict__ x, const float* __restrict__ t,
                                  const float* __restrict__ tW, const float* __restrict__ tb,
                                  _Float16* __restrict__ h16) {
  int n = blockIdx.x, j = threadIdx.x;
  float ta = t[n];
  float sv = sinf(ta * 1.57079632679f);
  float cv = cosf(ta * 1.57079632679f);
  float te = sv * tW[j] + cv * tW[256 + j] + ta * tW[512 + j] + tb[j];
  te = te / (1.f + expf(-te));                   // x*sigmoid(x)
  size_t base = (size_t)n * K0;
  h16[base + 128 + j] = (_Float16)te;
  if (j < 128) h16[base + j] = (_Float16)x[(size_t)n * 128 + j];
}

// ---------------- fuse Wl|Wr -> f16 in WMMA *fragment-major* layout + fused bias
// Bfrag[((nt*KB + kb)*32 + lane)*16 + j] = B[kb*32 + (lane>=16)*16 + j, nt*16 + (lane&15)]
// so each GEMM lane reads its 16 B halves as one contiguous 32-byte access.
__global__ void prep_weights_kernel(const float* __restrict__ Wl, const float* __restrict__ bl,
                                    const float* __restrict__ Wr, const float* __restrict__ br,
                                    _Float16* __restrict__ Bfrag, float* __restrict__ bct, int K) {
  int KB = K >> 5;
  int total = (F2 / 16) * KB * 32;
  int idx = blockIdx.x * blockDim.x + threadIdx.x;
  if (idx < F2) bct[idx] = (idx < F) ? bl[idx] : br[idx - F];
  if (idx >= total) return;
  int l     = idx & 31;
  int kb    = (idx >> 5) % KB;
  int nt    = (idx >> 5) / KB;
  int col   = nt * 16 + (l & 15);
  int krow0 = kb * 32 + (l >> 4) * 16;
  _Float16* o = Bfrag + (size_t)idx * 16;
#pragma unroll
  for (int j = 0; j < 16; ++j) {
    int k = krow0 + j;
    float v = (col < F) ? Wl[k * F + col] : Wr[k * F + (col - F)];
    o[j] = (_Float16)v;
  }
}

// ---------------- WMMA GEMM: C[M,N] = A[M,K](f16) @ B[K,N](f16,fragment-major) + bias
// one wave per 16x32 output tile (2 N-tiles share one A fragment), 4 waves (M-tiles)/block
__global__ void gemm_wmma_kernel(const _Float16* __restrict__ A, const v16h* __restrict__ Bf,
                                 const float* __restrict__ bias, float* __restrict__ C,
                                 int M, int K, int N) {
  int lane  = threadIdx.x & 31;
  int wave  = threadIdx.x >> 5;
  int mtile = blockIdx.x * 4 + wave;
  if (mtile * 16 >= M) return;
  int KB  = K >> 5;
  int nt0 = blockIdx.y * 2;
  int row = mtile * 16 + (lane & 15);
  int kif = (lane >> 4) * 16;           // A K sub-offset per half-wave (contiguous, matches Bfrag)
  const _Float16* Arow = A + (size_t)row * K;
  const v16h* B0 = Bf + (size_t)nt0 * KB * 32 + lane;
  const v16h* B1 = B0 + (size_t)KB * 32;
  v8f acc0 = {}, acc1 = {};
  for (int kb = 0; kb < KB; ++kb) {
    v8h alo = *(const v8h*)(Arow + kb * 32 + kif);        // K = kb*32+kif .. +7
    v8h ahi = *(const v8h*)(Arow + kb * 32 + kif + 8);    // K = kb*32+kif+8 .. +15
    v16h a;
#pragma unroll
    for (int j = 0; j < 8; ++j) { a[j] = alo[j]; a[8 + j] = ahi[j]; }
    v16h b0 = B0[kb * 32];
    v16h b1 = B1[kb * 32];
    acc0 = __builtin_amdgcn_wmma_f32_16x16x32_f16(false, a, false, b0, (short)0, acc0, false, false);
    acc1 = __builtin_amdgcn_wmma_f32_16x16x32_f16(false, a, false, b1, (short)0, acc1, false, false);
  }
  int rbase = mtile * 16 + (lane >> 4) * 8; // lanes 16-31 hold rows M=8..15 (C/D layout)
  int col0  = nt0 * 16 + (lane & 15);
  float bv0 = bias[col0], bv1 = bias[col0 + 16];
  float* Cp = C + (size_t)rbase * N + col0;
#pragma unroll
  for (int r = 0; r < 8; ++r) {
    Cp[(size_t)r * N]      = acc0[r] + bv0;
    Cp[(size_t)r * N + 16] = acc1[r] + bv1;
  }
}

// ---------------- per-layer node init: hout = out-bias, lmax = -inf, denom = 0
__global__ void node_init_kernel(const float* __restrict__ bias_out, float* __restrict__ hout,
                                 float* __restrict__ lmax, float* __restrict__ denom) {
  int idx = blockIdx.x * blockDim.x + threadIdx.x;
  if (idx >= N_NODES * F) return;
  int n = idx >> 7, c = idx & 127;
  hout[idx] = bias_out[c];
  if (c < HEADS) { lmax[n * HEADS + c] = -3.0e38f; denom[n * HEADS + c] = 0.f; }
}

// ---------------- pass 1: logits + segment-max (one thread per edge*head)
__global__ void edge_logits_kernel(const int* __restrict__ ei, const float* __restrict__ xlr,
                                   const float* __restrict__ att, float* __restrict__ elog,
                                   float* __restrict__ lmax) {
  int idx = blockIdx.x * blockDim.x + threadIdx.x;
  if (idx >= ETOT * HEADS) return;
  int e = idx >> 2, h = idx & 3;
  int s, d; edge_sd(ei, e, s, d);
  const float4* xl4 = (const float4*)(xlr + (size_t)s * F2 + h * HID);
  const float4* xr4 = (const float4*)(xlr + (size_t)d * F2 + F + h * HID);
  const float4* at4 = (const float4*)(att + h * HID);
  float lg = 0.f;
#pragma unroll
  for (int i = 0; i < 8; ++i) {
    float4 a = xl4[i], b = xr4[i], w = at4[i];
    lg += lrelu(a.x + b.x) * w.x + lrelu(a.y + b.y) * w.y +
          lrelu(a.z + b.z) * w.z + lrelu(a.w + b.w) * w.w;
  }
  elog[idx] = lg;
  atomicMaxF(&lmax[d * HEADS + h], lg);
}

// ---------------- pass 2: e = exp(logit - max), segment-sum denom
__global__ void edge_exp_kernel(const int* __restrict__ ei, float* __restrict__ elog,
                                const float* __restrict__ lmax, float* __restrict__ denom) {
  int idx = blockIdx.x * blockDim.x + threadIdx.x;
  if (idx >= ETOT * HEADS) return;
  int e = idx >> 2, h = idx & 3;
  int s, d; edge_sd(ei, e, s, d);
  float ev = expf(elog[idx] - lmax[d * HEADS + h]);
  elog[idx] = ev;
  atomicAdd(&denom[d * HEADS + h], ev);
}

// ---------------- pass 3: hout[dst] += alpha * xl[src]
__global__ void edge_scatter_kernel(const int* __restrict__ ei, const float* __restrict__ elog,
                                    const float* __restrict__ denom, const float* __restrict__ xlr,
                                    float* __restrict__ hout) {
  int idx = blockIdx.x * blockDim.x + threadIdx.x;
  if (idx >= ETOT * HEADS) return;
  int e = idx >> 2, h = idx & 3;
  int s, d; edge_sd(ei, e, s, d);
  float alpha = elog[idx] / (denom[d * HEADS + h] + 1e-16f);
  const float* xl = xlr + (size_t)s * F2 + h * HID;
  float* op = hout + (size_t)d * F + h * HID;
#pragma unroll
  for (int c = 0; c < HID; ++c) atomicAdd(&op[c], alpha * xl[c]);
}

// ---------------- graph layernorm: gst = [sum(64) | sumsq(64) | count(64)]
__global__ void zero_gstats_kernel(float* g) { if (threadIdx.x < 192) g[threadIdx.x] = 0.f; }

__global__ void ln_stats_kernel(const float* __restrict__ hout, const int* __restrict__ batch,
                                float* __restrict__ gst) {
  int n = blockIdx.x * blockDim.x + threadIdx.x;
  if (n >= N_NODES) return;
  const float4* hp = (const float4*)(hout + (size_t)n * F);
  float s = 0.f, ss = 0.f;
#pragma unroll
  for (int i = 0; i < 32; ++i) {
    float4 v = hp[i];
    s  += v.x + v.y + v.z + v.w;
    ss += v.x * v.x + v.y * v.y + v.z * v.z + v.w * v.w;
  }
  int g = batch[n];
  atomicAdd(&gst[g], s);
  atomicAdd(&gst[64 + g], ss);
  atomicAdd(&gst[128 + g], (float)F);
}

__global__ void ln_final_kernel(float* gst) {
  int g = threadIdx.x;
  if (g >= N_GRAPHS) return;
  float cnt  = fmaxf(gst[128 + g], 1.f);
  float mean = gst[g] / cnt;
  float var  = gst[64 + g] / cnt - mean * mean;   // E[h^2] - mean^2
  gst[g]      = mean;
  gst[64 + g] = rsqrtf(var + LN_EPS);
}

// normalize, write f32 h (for edges/link) and packed f16 copy (next layer's GEMM A)
__global__ void ln_apply_kernel(const float* __restrict__ hout, const int* __restrict__ batch,
                                const float* __restrict__ gst, const float* __restrict__ w,
                                const float* __restrict__ b, float* __restrict__ h,
                                _Float16* __restrict__ h16) {
  int idx = blockIdx.x * blockDim.x + threadIdx.x;
  if (idx >= N_NODES * F) return;
  int n = idx >> 7, c = idx & 127;
  int g = batch[n];
  float v = (hout[idx] - gst[g]) * gst[64 + g] * w[c] + b[c];
  h[idx]   = v;
  h16[idx] = (_Float16)v;
}

// ---------------- link scoring: sigmoid((h[a]*h[b]) @ link_W + link_b)
__global__ void link_kernel(const int* __restrict__ pi, const float* __restrict__ h,
                            const float* __restrict__ lW, const float* __restrict__ lb,
                            float* __restrict__ out) {
  int p = blockIdx.x * blockDim.x + threadIdx.x;
  if (p >= N_PAIRS) return;
  int a = pi[p], b = pi[N_PAIRS + p];
  const float4* ha = (const float4*)(h + (size_t)a * F);
  const float4* hb = (const float4*)(h + (size_t)b * F);
  const float4* w4 = (const float4*)lW;
  float acc = 0.f;
#pragma unroll
  for (int i = 0; i < 32; ++i) {
    float4 u = ha[i], v = hb[i], w = w4[i];
    acc += u.x * v.x * w.x + u.y * v.y * w.y + u.z * v.z * w.z + u.w * v.w * w.w;
  }
  acc += lb[0];
  out[p] = 1.f / (1.f + expf(-acc));
}

extern "C" void kernel_launch(void* const* d_in, const int* in_sizes, int n_in,
                              void* d_out, int out_size, void* d_ws, size_t ws_size,
                              hipStream_t stream) {
  const float* x     = (const float*)d_in[0];
  const float* t     = (const float*)d_in[1];
  const int*   ei    = (const int*)d_in[2];
  const int*   pi    = (const int*)d_in[3];
  const int*   batch = (const int*)d_in[4];
  const float* tW    = (const float*)d_in[5];
  const float* tb    = (const float*)d_in[6];
  const float* W0l   = (const float*)d_in[7];
  const float* b0l   = (const float*)d_in[8];
  const float* W0r   = (const float*)d_in[9];
  const float* b0r   = (const float*)d_in[10];
  const float* att0  = (const float*)d_in[11];
  const float* bias0 = (const float*)d_in[12];
  const float* Wl    = (const float*)d_in[13];
  const float* bl    = (const float*)d_in[14];
  const float* Wr    = (const float*)d_in[15];
  const float* br    = (const float*)d_in[16];
  const float* att   = (const float*)d_in[17];
  const float* bias  = (const float*)d_in[18];
  const float* lnw   = (const float*)d_in[19];
  const float* lnb   = (const float*)d_in[20];
  const float* lW    = (const float*)d_in[21];
  const float* lb    = (const float*)d_in[22];
  float* out = (float*)d_out;
  (void)in_sizes; (void)n_in; (void)out_size; (void)ws_size;

  char* wsb = (char*)d_ws;
  size_t off = 0;
  auto alloc = [&](size_t bytes) -> char* {
    char* p = wsb + off; off += (bytes + 255) & ~(size_t)255; return p;
  };
  _Float16* h16  = (_Float16*)alloc((size_t)N_NODES * K0 * 2);   // f16 GEMM input (stride K)
  float*    xlr  = (float*)   alloc((size_t)N_NODES * F2 * 4);   // xl|xr fused
  float*    hout = (float*)   alloc((size_t)N_NODES * F * 4);    // aggregation output
  float*    hbuf = (float*)   alloc((size_t)N_NODES * F * 4);    // normalized features
  float*    elog = (float*)   alloc((size_t)ETOT * HEADS * 4);   // logits -> exp values
  float*    lmax = (float*)   alloc((size_t)N_NODES * HEADS * 4);
  float*    den  = (float*)   alloc((size_t)N_NODES * HEADS * 4);
  _Float16* Bcat = (_Float16*)alloc((size_t)K0 * F2 * 2);        // fused f16 weights (fragment-major)
  float*    bct  = (float*)   alloc((size_t)F2 * 4);
  float*    gst  = (float*)   alloc((size_t)192 * 4);

  time_embed_kernel<<<N_NODES, 256, 0, stream>>>(x, t, tW, tb, h16);

  const int NE4 = (ETOT * HEADS + 255) / 256;
  const int NF  = (N_NODES * F + 255) / 256;
  const int MT  = (N_NODES + 15) / 16;          // 3125 M-tiles

  for (int L = 0; L < 4; ++L) {
    int K = (L == 0) ? K0 : F;
    const float* Wlp = (L == 0) ? W0l  : Wl  + (size_t)(L - 1) * F * F;
    const float* blp = (L == 0) ? b0l  : bl  + (size_t)(L - 1) * F;
    const float* Wrp = (L == 0) ? W0r  : Wr  + (size_t)(L - 1) * F * F;
    const float* brp = (L == 0) ? b0r  : br  + (size_t)(L - 1) * F;
    const float* atp = (L == 0) ? att0 : att + (size_t)(L - 1) * HEADS * HID;
    const float* bop = (L == 0) ? bias0 : bias + (size_t)(L - 1) * F;

    int pthreads = (F2 / 16) * (K >> 5) * 32;   // one thread per (ntile, kblock, lane)
    prep_weights_kernel<<<(pthreads + 255) / 256, 256, 0, stream>>>(Wlp, blp, Wrp, brp,
                                                                    Bcat, bct, K);
    dim3 gg((MT + 3) / 4, F2 / 32);             // 2 N-tiles per wave
    gemm_wmma_kernel<<<gg, 128, 0, stream>>>(h16, (const v16h*)Bcat, bct, xlr, N_NODES, K, F2);
    node_init_kernel<<<NF, 256, 0, stream>>>(bop, hout, lmax, den);
    edge_logits_kernel<<<NE4, 256, 0, stream>>>(ei, xlr, atp, elog, lmax);
    edge_exp_kernel<<<NE4, 256, 0, stream>>>(ei, elog, lmax, den);
    edge_scatter_kernel<<<NE4, 256, 0, stream>>>(ei, elog, den, xlr, hout);
    zero_gstats_kernel<<<1, 192, 0, stream>>>(gst);
    ln_stats_kernel<<<(N_NODES + 255) / 256, 256, 0, stream>>>(hout, batch, gst);
    ln_final_kernel<<<1, 64, 0, stream>>>(gst);
    ln_apply_kernel<<<NF, 256, 0, stream>>>(hout, batch, gst,
                                            lnw + (size_t)L * F, lnb + (size_t)L * F, hbuf, h16);
  }
  link_kernel<<<(N_PAIRS + 255) / 256, 256, 0, stream>>>(pi, hbuf, lW, lb, out);
}
